// PointNet2_85461259256572
// MI455X (gfx1250) — compile-verified, matrix operand order checked by reading, not run on hardware
//
#include <hip/hip_runtime.h>
#include <hip/hip_bf16.h>

typedef __attribute__((ext_vector_type(16))) _Float16 v16h;
typedef __attribute__((ext_vector_type(8)))  _Float16 v8h;
typedef __attribute__((ext_vector_type(8)))  float    v8f;

#define EPS_BN 1e-5f

// ---------------------------------------------------------------------------
// Fused GEMM + BatchNorm + ReLU using v_wmma_f32_16x16x32_f16.
// X:  [R, Kp] f16 row-major, K padded to multiple of 32, pad = 0
// Wh: [Cout, Kp] f16 row-major (so B[k][n] = Wh[n][k]), pad = 0
// Y:  [R, Cout] -- f16 if out16 (feeds next GEMM) else f32
// y = relu(scale[n]*acc + shift[n]); scale = gamma*rsqrt(rv+eps),
// shift = beta + scale*(b - rm)
// Each wave: 16(M) x 32(N) tile, 2 accumulators, A fragment reused.
// grid.x = R/128 (8 waves stacked in M per block), grid.y = Cout/32.
// Hot loop: 6x global_load_b128 + 2x wmma, branch-free.
// ---------------------------------------------------------------------------
__launch_bounds__(256)
__global__ void gemm_bn_relu(const _Float16* __restrict__ X, const _Float16* __restrict__ Wh,
                             const float* __restrict__ Bb, const float* __restrict__ Gm,
                             const float* __restrict__ Bt, const float* __restrict__ Rm,
                             const float* __restrict__ Rv,
                             void* __restrict__ Yv, int R, int Kp, int Cout, int out16) {
  const int lane = threadIdx.x & 31;
  const int wave = threadIdx.x >> 5;
  const int half = lane >> 4;   // K/N half of the wave
  const int l    = lane & 15;
  const int m0   = (blockIdx.x * 8 + wave) * 16;
  const int n0   = blockIdx.y * 32;

  const int arow  = m0 + l;          // A: lane holds row M = l
  const int bcol0 = n0 + l;          // B tile 0: lane holds col N = l
  const int bcol1 = n0 + 16 + l;     // B tile 1

  // ISA 7.12.2 16-bit fragment: lanes 0-15 hold K {k0..k0+7, k0+16..k0+23},
  // lanes 16-31 hold K {k0+8..k0+15, k0+24..k0+31}; 2 packed f16 per VGPR.
  const _Float16* xr  = X  + (size_t)arow  * Kp + half * 8;
  const _Float16* w0p = Wh + (size_t)bcol0 * Kp + half * 8;
  const _Float16* w1p = Wh + (size_t)bcol1 * Kp + half * 8;

  v8f c0 = {};
  v8f c1 = {};

  for (int k0 = 0; k0 < Kp; k0 += 32) {
    v8h alo  = *(const v8h*)(xr  + k0);
    v8h ahi  = *(const v8h*)(xr  + k0 + 16);
    v8h b0lo = *(const v8h*)(w0p + k0);
    v8h b0hi = *(const v8h*)(w0p + k0 + 16);
    v8h b1lo = *(const v8h*)(w1p + k0);
    v8h b1hi = *(const v8h*)(w1p + k0 + 16);
    v16h a  = __builtin_shufflevector(alo,  ahi,  0,1,2,3,4,5,6,7,8,9,10,11,12,13,14,15);
    v16h b0 = __builtin_shufflevector(b0lo, b0hi, 0,1,2,3,4,5,6,7,8,9,10,11,12,13,14,15);
    v16h b1 = __builtin_shufflevector(b1lo, b1hi, 0,1,2,3,4,5,6,7,8,9,10,11,12,13,14,15);
    c0 = __builtin_amdgcn_wmma_f32_16x16x32_f16(false, a, false, b0, (short)0, c0, false, false);
    c1 = __builtin_amdgcn_wmma_f32_16x16x32_f16(false, a, false, b1, (short)0, c1, false, false);
  }

  // Fused BN + ReLU epilogue.
  float s0 = Gm[bcol0] * rsqrtf(Rv[bcol0] + EPS_BN);
  float o0 = Bt[bcol0] + s0 * (Bb[bcol0] - Rm[bcol0]);
  float s1 = Gm[bcol1] * rsqrtf(Rv[bcol1] + EPS_BN);
  float o1 = Bt[bcol1] + s1 * (Bb[bcol1] - Rm[bcol1]);
  if (out16) {
    _Float16* Y = (_Float16*)Yv;
#pragma unroll
    for (int v = 0; v < 8; ++v) {
      // C/D layout: lane = N (+16 selects M>=8 half), vgpr v = M & 7
      int m = m0 + v + half * 8;
      float y0 = s0 * c0[v] + o0; y0 = y0 > 0.f ? y0 : 0.f;
      float y1 = s1 * c1[v] + o1; y1 = y1 > 0.f ? y1 : 0.f;
      Y[(size_t)m * Cout + bcol0] = (_Float16)y0;
      Y[(size_t)m * Cout + bcol1] = (_Float16)y1;
    }
  } else {
    float* Y = (float*)Yv;
#pragma unroll
    for (int v = 0; v < 8; ++v) {
      int m = m0 + v + half * 8;
      float y0 = s0 * c0[v] + o0; y0 = y0 > 0.f ? y0 : 0.f;
      float y1 = s1 * c1[v] + o1; y1 = y1 > 0.f ? y1 : 0.f;
      Y[(size_t)m * Cout + bcol0] = y0;
      Y[(size_t)m * Cout + bcol1] = y1;
    }
  }
}

// Pre-convert weights to padded f16: Wh[n*Kp + k] = k < K ? (f16)W[n*K+k] : 0
__global__ void convert_w_kernel(const float* __restrict__ W, _Float16* __restrict__ Wh,
                                 int K, int Kp, int total) {
  int t = blockIdx.x * blockDim.x + threadIdx.x;
  if (t >= total) return;               // total = Cout*Kp
  int k = t % Kp;
  int n = t / Kp;
  Wh[t] = (k < K) ? (_Float16)W[(size_t)n * K + k] : (_Float16)0.f;
}

// ---------------------------------------------------------------------------
// Farthest point sampling: one workgroup per batch, distance array in LDS.
// ---------------------------------------------------------------------------
__launch_bounds__(256)
__global__ void fps_kernel(const float* __restrict__ xyz, int N, int npoint,
                           int* __restrict__ out) {
  __shared__ float dist[4096];
  __shared__ float rv[256];
  __shared__ int   ri[256];
  const int tid = threadIdx.x;
  const int b   = blockIdx.x;
  const float* p = xyz + (size_t)b * N * 3;
  for (int i = tid; i < N; i += 256) dist[i] = 1e10f;
  __syncthreads();
  int far = 0;
  for (int s = 0; s < npoint; ++s) {
    if (tid == 0) out[b * npoint + s] = far;
    float cx = p[far * 3 + 0], cy = p[far * 3 + 1], cz = p[far * 3 + 2];
    float best = -1.f; int bi = 0;
    for (int i = tid; i < N; i += 256) {
      float dx = p[i * 3 + 0] - cx;
      float dy = p[i * 3 + 1] - cy;
      float dz = p[i * 3 + 2] - cz;
      float d  = dx * dx + dy * dy + dz * dz;
      float m  = dist[i];
      m = d < m ? d : m;
      dist[i] = m;
      if (m > best) { best = m; bi = i; }
    }
    rv[tid] = best; ri[tid] = bi;
    __syncthreads();
    for (int off = 128; off > 0; off >>= 1) {
      if (tid < off) {
        float ov = rv[tid + off]; int oi = ri[tid + off];
        if (ov > rv[tid] || (ov == rv[tid] && oi < ri[tid])) { rv[tid] = ov; ri[tid] = oi; }
      }
      __syncthreads();
    }
    far = ri[0];
    __syncthreads();
  }
}

// new_xyz[b,s,:] = xyz[b, fidx[b,s], :]
__global__ void gather_xyz_kernel(const float* __restrict__ xyz, const int* __restrict__ fidx,
                                  float* __restrict__ out, int N, int S, int total) {
  int t = blockIdx.x * blockDim.x + threadIdx.x;
  if (t >= total) return;               // total = B*S
  int b = t / S;
  int i = fidx[t];
  const float* src = xyz + ((size_t)b * N + i) * 3;
  out[(size_t)t * 3 + 0] = src[0];
  out[(size_t)t * 3 + 1] = src[1];
  out[(size_t)t * 3 + 2] = src[2];
}

// Ball query, first-nsample-in-index-order (matches reference sort trick).
__global__ void ball_query_kernel(const float* __restrict__ xyz, const float* __restrict__ nxyz,
                                  int N, int S, int nsample, float r2,
                                  int* __restrict__ idx, int total) {
  int t = blockIdx.x * blockDim.x + threadIdx.x;
  if (t >= total) return;               // total = B*S
  int b = t / S;
  const float* q = nxyz + (size_t)t * 3;
  float qx = q[0], qy = q[1], qz = q[2];
  const float* p = xyz + (size_t)b * N * 3;
  int* o = idx + (size_t)t * nsample;
  int cnt = 0, first = 0;
  for (int i = 0; i < N && cnt < nsample; ++i) {
    float dx = p[i * 3 + 0] - qx;
    float dy = p[i * 3 + 1] - qy;
    float dz = p[i * 3 + 2] - qz;
    float d  = dx * dx + dy * dy + dz * dz;
    if (d <= r2) {
      if (cnt == 0) first = i;
      o[cnt++] = i;
    }
  }
  for (; cnt < nsample; ++cnt) o[cnt] = first;
}

// Grouped features -> f16, row stride Kp, zero-padded:
// X[row, 0:3] = xyz[idx]-center, X[row, 3:3+Cpts] = pts[idx], X[row, 3+Cpts:Kp] = 0
__global__ void group_kernel(const float* __restrict__ xyz, const float* __restrict__ nxyz,
                             const float* __restrict__ pts, const int* __restrict__ idx,
                             _Float16* __restrict__ X, int N, int S, int nsample,
                             int Cpts, int Kp, int total) {
  int t = blockIdx.x * blockDim.x + threadIdx.x;
  if (t >= total) return;               // total = B*S*nsample
  int j  = t % nsample;
  int bs = t / nsample;
  int b  = bs / S;
  int i  = idx[(size_t)bs * nsample + j];
  const float* src = xyz + ((size_t)b * N + i) * 3;
  const float* c   = nxyz + (size_t)bs * 3;
  _Float16* row = X + (size_t)t * Kp;
  row[0] = (_Float16)(src[0] - c[0]);
  row[1] = (_Float16)(src[1] - c[1]);
  row[2] = (_Float16)(src[2] - c[2]);
  const float* ps = pts + ((size_t)b * N + i) * Cpts;
  for (int k = 0; k < Cpts; ++k) row[3 + k] = (_Float16)ps[k];
  for (int k = 3 + Cpts; k < Kp; ++k) row[k] = (_Float16)0.f;
}

// Max over nsample: out[g, c] = max_j X[g*nsample + j, c]   (f32 in/out)
__global__ void maxpool_kernel(const float* __restrict__ X, float* __restrict__ out,
                               int nsample, int C, int total) {
  int t = blockIdx.x * blockDim.x + threadIdx.x;
  if (t >= total) return;               // total = B*S*C
  int c = t % C;
  int g = t / C;
  float m = -3.4e38f;
  for (int j = 0; j < nsample; ++j) {
    float v = X[((size_t)g * nsample + j) * C + c];
    m = v > m ? v : m;
  }
  out[t] = m;
}

// 3-NN + inverse-distance weights.
__global__ void knn3_kernel(const float* __restrict__ xyz1, const float* __restrict__ xyz2,
                            int N1, int N2, int* __restrict__ idx3, float* __restrict__ w3,
                            int total) {
  int t = blockIdx.x * blockDim.x + threadIdx.x;
  if (t >= total) return;               // total = B*N1
  int b = t / N1;
  const float* q = xyz1 + (size_t)t * 3;
  float qx = q[0], qy = q[1], qz = q[2];
  const float* p = xyz2 + (size_t)b * N2 * 3;
  float d0 = 3.4e38f, d1 = 3.4e38f, d2 = 3.4e38f;
  int   i0 = 0, i1 = 0, i2 = 0;
  for (int i = 0; i < N2; ++i) {
    float dx = p[i * 3 + 0] - qx;
    float dy = p[i * 3 + 1] - qy;
    float dz = p[i * 3 + 2] - qz;
    float d  = dx * dx + dy * dy + dz * dz;
    if (d < d0)      { d2 = d1; i2 = i1; d1 = d0; i1 = i0; d0 = d; i0 = i; }
    else if (d < d1) { d2 = d1; i2 = i1; d1 = d;  i1 = i; }
    else if (d < d2) { d2 = d;  i2 = i; }
  }
  float w0 = 1.f / (d0 + 1e-8f), w1 = 1.f / (d1 + 1e-8f), w2 = 1.f / (d2 + 1e-8f);
  float ws = w0 + w1 + w2;
  idx3[t * 3 + 0] = i0; idx3[t * 3 + 1] = i1; idx3[t * 3 + 2] = i2;
  w3[t * 3 + 0] = w0 / ws; w3[t * 3 + 1] = w1 / ws; w3[t * 3 + 2] = w2 / ws;
}

// X[row, 0:C1] = pts1[row], X[row, C1:C] = sum_k w[k]*pts2[idx[k]]  (f16 out)
// C = C1+C2 is always a multiple of 32 here, so no padding needed.
__global__ void interp_concat_kernel(const float* __restrict__ pts1, const float* __restrict__ pts2,
                                     const int* __restrict__ idx3, const float* __restrict__ w3,
                                     _Float16* __restrict__ X, int C1, int C2, int N1, int N2,
                                     int total) {
  int t = blockIdx.x * blockDim.x + threadIdx.x;
  if (t >= total) return;               // total = B*N1*(C1+C2)
  int C   = C1 + C2;
  int rc  = t % C;
  int row = t / C;
  int b   = row / N1;
  if (rc < C1) {
    X[t] = (_Float16)pts1[(size_t)row * C1 + rc];
  } else {
    int c = rc - C1;
    const int*   id = idx3 + (size_t)row * 3;
    const float* w  = w3   + (size_t)row * 3;
    const float* p  = pts2 + (size_t)b * N2 * C2;
    X[t] = (_Float16)(w[0] * p[(size_t)id[0] * C2 + c]
                    + w[1] * p[(size_t)id[1] * C2 + c]
                    + w[2] * p[(size_t)id[2] * C2 + c]);
  }
}

// in[(b*C + c)*N + n] -> out[(b*N + n)*C + c]
__global__ void transpose_in_kernel(const float* __restrict__ in, float* __restrict__ out,
                                    int C, int N, int total) {
  int t = blockIdx.x * blockDim.x + threadIdx.x;
  if (t >= total) return;               // total = B*C*N
  int n  = t % N;
  int bc = t / N;
  int c  = bc % C;
  int b  = bc / C;
  out[((size_t)b * N + n) * C + c] = in[t];
}

// x[(b*N + n)*C + c] -> out[(b*C + c)*N + n]
__global__ void transpose_out_kernel(const float* __restrict__ x, float* __restrict__ out,
                                     int N, int C, int total) {
  int t = blockIdx.x * blockDim.x + threadIdx.x;
  if (t >= total) return;               // total = B*N*C
  int c  = t % C;
  int bn = t / C;
  int n  = bn % N;
  int b  = bn / N;
  out[((size_t)b * C + c) * N + n] = x[t];
}

// ---------------------------------------------------------------------------
// Host orchestration
// ---------------------------------------------------------------------------
struct Layer {
  const float *w, *b, *gamma, *beta, *rm, *rv;
  _Float16* wh;   // padded f16 weights in workspace
  int cin, kp, cout;
};

static inline int cdiv(int a, int b) { return (a + b - 1) / b; }

static void launch_gemm(const void* X, const Layer& L, void* Y, int R, int out16,
                        hipStream_t st) {
  dim3 grid(R / 128, L.cout / 32);
  gemm_bn_relu<<<grid, 256, 0, st>>>((const _Float16*)X, L.wh, L.b, L.gamma, L.beta,
                                     L.rm, L.rv, Y, R, L.kp, L.cout, out16);
}

// Set-abstraction stage. Final pooled f32 features -> outfeat [B*S, Cout_last].
static void run_sa(const float* xyz, const float* pts, int B, int N, int Cpts,
                   int S, float radius, int nsample, const Layer* Ls, int nl,
                   float* newxyz, float* outfeat, int* fidx, int* bidx,
                   void* ping, void* pong, hipStream_t st) {
  fps_kernel<<<B, 256, 0, st>>>(xyz, N, S, fidx);
  int tS = B * S;
  gather_xyz_kernel<<<cdiv(tS, 256), 256, 0, st>>>(xyz, fidx, newxyz, N, S, tS);
  ball_query_kernel<<<cdiv(tS, 256), 256, 0, st>>>(xyz, newxyz, N, S, nsample,
                                                   radius * radius, bidx, tS);
  int R = B * S * nsample;
  int Kp0 = Ls[0].kp;
  group_kernel<<<cdiv(R, 256), 256, 0, st>>>(xyz, newxyz, pts, bidx, (_Float16*)ping,
                                             N, S, nsample, Cpts, Kp0, R);
  const void* cur = ping;
  void* buf[2] = {pong, ping};
  for (int i = 0; i < nl; ++i) {
    void* dst = buf[i & 1];
    launch_gemm(cur, Ls[i], dst, R, (i < nl - 1) ? 1 : 0, st);   // final layer f32
    cur = dst;
  }
  int C = Ls[nl - 1].cout;
  int total = B * S * C;
  maxpool_kernel<<<cdiv(total, 256), 256, 0, st>>>((const float*)cur, outfeat,
                                                   nsample, C, total);
}

// Feature-propagation stage; final layer -> outfeat (f32 unless final16).
static void run_fp(const float* xyz1, const float* xyz2, const float* pts1, const float* pts2,
                   int B, int N1, int N2, int C1, int C2, const Layer* Ls, int nl,
                   int* idx3, float* w3, void* ping, void* pong, void* outfeat,
                   int final16, hipStream_t st) {
  int tN = B * N1;
  knn3_kernel<<<cdiv(tN, 256), 256, 0, st>>>(xyz1, xyz2, N1, N2, idx3, w3, tN);
  int C = C1 + C2;
  int total = B * N1 * C;
  interp_concat_kernel<<<cdiv(total, 256), 256, 0, st>>>(pts1, pts2, idx3, w3,
                                                         (_Float16*)ping, C1, C2, N1, N2, total);
  int R = B * N1;
  const void* cur = ping;
  void* buf[2] = {pong, ping};
  for (int i = 0; i < nl; ++i) {
    int last = (i == nl - 1);
    void* dst = last ? outfeat : buf[i & 1];
    launch_gemm(cur, Ls[i], dst, R, last ? final16 : 1, st);
    cur = dst;
  }
}

extern "C" void kernel_launch(void* const* d_in, const int* in_sizes, int n_in,
                              void* d_out, int out_size, void* d_ws, size_t ws_size,
                              hipStream_t stream) {
  (void)in_sizes; (void)n_in; (void)out_size; (void)ws_size;
  const int B = 8, N0 = 4096;

  // --- parameter table: setup_inputs() dict order -------------------------
  static const int LDIMS[22][2] = {
    {6, 32}, {32, 32}, {32, 64},          // sa1
    {67, 64}, {64, 64}, {64, 128},        // sa2
    {131, 128}, {128, 128}, {128, 256},   // sa3
    {259, 256}, {256, 256}, {256, 512},   // sa4
    {768, 256}, {256, 256},               // fp4
    {384, 256}, {256, 256},               // fp3
    {320, 256}, {256, 128},               // fp2
    {128, 128}, {128, 128}, {128, 128},   // fp1
    {128, 128}                            // head
  };
  const float* inputs = (const float*)d_in[0];
  Layer L[22];
  {
    int p = 1;
    for (int i = 0; i < 22; ++i) {
      L[i].w     = (const float*)d_in[p++];
      L[i].b     = (const float*)d_in[p++];
      L[i].gamma = (const float*)d_in[p++];
      L[i].beta  = (const float*)d_in[p++];
      L[i].rm    = (const float*)d_in[p++];
      L[i].rv    = (const float*)d_in[p++];
      L[i].cin  = LDIMS[i][0];
      L[i].kp   = (LDIMS[i][0] + 31) & ~31;
      L[i].cout = LDIMS[i][1];
    }
  }

  // --- deterministic workspace bump allocator -----------------------------
  char* wptr = (char*)d_ws;
  auto alloc_b = [&](size_t bytes) { void* p = wptr; wptr += ((bytes + 255) & ~(size_t)255); return p; };
  auto alloc_f = [&](size_t n) { return (float*)alloc_b(n * 4); };
  auto alloc_i = [&](size_t n) { return (int*)alloc_b(n * 4); };
  auto alloc_h = [&](size_t n) { return (_Float16*)alloc_b(n * 2); };

  float* coords = alloc_f((size_t)B * N0 * 3);
  float* l1x = alloc_f((size_t)B * 1024 * 3);
  float* l2x = alloc_f((size_t)B * 256 * 3);
  float* l3x = alloc_f((size_t)B * 64 * 3);
  float* l4x = alloc_f((size_t)B * 16 * 3);
  float* l1f = alloc_f((size_t)B * 1024 * 128);  // 64ch after sa1, 128ch after fp2
  float* l2f = alloc_f((size_t)B * 256 * 256);   // 128ch after sa2, 256ch after fp3
  float* l3f = alloc_f((size_t)B * 64 * 256);    // 256ch after sa3, 256ch after fp4
  float* l4f = alloc_f((size_t)B * 16 * 512);
  int*   fidx = alloc_i((size_t)B * 1024);
  int*   bidx = alloc_i((size_t)B * 1024 * 32);
  int*   idx3 = alloc_i((size_t)B * N0 * 3);
  float* w3   = alloc_f((size_t)B * N0 * 3);
  void*  ping = alloc_b((size_t)262144 * 64 * 4);
  void*  pong = alloc_b((size_t)262144 * 64 * 4);
  for (int i = 0; i < 22; ++i) L[i].wh = alloc_h((size_t)L[i].cout * L[i].kp);

  // --- weight pre-conversion to padded f16 --------------------------------
  for (int i = 0; i < 22; ++i) {
    int total = L[i].cout * L[i].kp;
    convert_w_kernel<<<cdiv(total, 256), 256, 0, stream>>>(L[i].w, L[i].wh,
                                                           L[i].cin, L[i].kp, total);
  }

  // --- stage 0: inputs (B,3,N) -> coords (B,N,3); l0 features == coords ---
  {
    int total = B * 3 * N0;
    transpose_in_kernel<<<cdiv(total, 256), 256, 0, stream>>>(inputs, coords, 3, N0, total);
  }

  // --- set-abstraction tower ---------------------------------------------
  run_sa(coords, coords, B, N0,   3, 1024, 0.1f, 32, &L[0], 3, l1x, l1f, fidx, bidx, ping, pong, stream);
  run_sa(l1x,    l1f,    B, 1024, 64, 256,  0.2f, 32, &L[3], 3, l2x, l2f, fidx, bidx, ping, pong, stream);
  run_sa(l2x,    l2f,    B, 256, 128, 64,   0.4f, 32, &L[6], 3, l3x, l3f, fidx, bidx, ping, pong, stream);
  run_sa(l3x,    l3f,    B, 64,  256, 16,   0.8f, 32, &L[9], 3, l4x, l4f, fidx, bidx, ping, pong, stream);

  // --- feature-propagation tower (each overwrites the finer-level features) ---
  run_fp(l3x, l4x, l3f, l4f, B, 64,   16,  256, 512, &L[12], 2, idx3, w3, ping, pong, l3f, 0, stream);
  run_fp(l2x, l3x, l2f, l3f, B, 256,  64,  128, 256, &L[14], 2, idx3, w3, ping, pong, l2f, 0, stream);
  run_fp(l1x, l2x, l1f, l2f, B, 1024, 256, 64,  256, &L[16], 2, idx3, w3, ping, pong, l1f, 0, stream);
  // fp1: points1 = None -> C1 = 0; 3 layers, final f16 lands in pong (feeds head)
  run_fp(coords, l1x, nullptr, l1f, B, N0, 1024, 0, 128, &L[18], 3, idx3, w3, ping, pong, pong, 1, stream);

  // --- head: conv_bn_relu on pong(f16) -> ping(f32) -----------------------
  launch_gemm(pong, L[21], ping, B * N0, 0, stream);

  // --- outputs: (x^T [8,128,4096], l4^T [8,512,16]) ----------------------
  float* out = (float*)d_out;
  {
    int total = B * N0 * 128;
    transpose_out_kernel<<<cdiv(total, 256), 256, 0, stream>>>((const float*)ping, out,
                                                               N0, 128, total);
  }
  {
    int total = B * 16 * 512;
    transpose_out_kernel<<<cdiv(total, 256), 256, 0, stream>>>(l4f, out + (size_t)B * 128 * N0,
                                                               16, 512, total);
  }
}
